// MultiHeadAttention_79156247265899
// MI455X (gfx1250) — compile-verified, matrix operand order checked by reading, not run on hardware
//
#include <hip/hip_runtime.h>
#include <hip/hip_bf16.h>
#include <math.h>

#define D_MODEL 1024
#define N_HEADS 16
#define D_HEAD  64
#define NQ      2048
#define NK      2048
#define BATCH   2
#define LN_EPS  1e-5f

typedef __attribute__((ext_vector_type(16))) __bf16 v16bf;
typedef __attribute__((ext_vector_type(8)))  float  v8f;

// CDNA5 16-bit A/B fragment K-index for lane/element (ISA 7.12.2):
// lanes 0-15 hold K={0..7,16..23}, lanes 16-31 hold K={8..15,24..31}
__device__ __forceinline__ int frag_k(int lane, int e) {
  return ((lane >> 4) << 3) + e + ((e >= 8) ? 8 : 0);
}

__device__ __forceinline__ v8f wmma_bf16(v16bf a, v16bf b, v8f c) {
  return __builtin_amdgcn_wmma_f32_16x16x32_bf16(false, a, false, b,
                                                 (short)0, c, false, false);
}

// CDNA5 async global->LDS copy (ASYNCcnt-tracked). Flat LDS addresses
// truncate to the wave-relative LDS byte offset (ISA 10.2 aperture rules).
__device__ __forceinline__ void async_b128(void* lds, const void* g) {
  unsigned l = (unsigned)(size_t)lds;
  asm volatile("global_load_async_to_lds_b128 %0, %1, off"
               :: "v"(l), "v"(g) : "memory");
}

// ---------------------------------------------------------------------------
// QKV projection: Y = X @ W^T + b, output head-split bf16 [B, H, Nseq, 64]
// block = 256 threads (8 waves), tile = 128 rows x 64 cols, K staged 32-wide.
// ---------------------------------------------------------------------------
__global__ __launch_bounds__(256)
void proj_qkv_kernel(const float* __restrict__ X, const float* __restrict__ W,
                     const float* __restrict__ bias, __bf16* __restrict__ out) {
  __shared__ __bf16 Xs[128 * 32];
  __shared__ __bf16 Ws[64 * 32];
  const int tid  = threadIdx.x;
  const int wave = tid >> 5, lane = tid & 31;
  const int lrow = lane & 15, lhalf = lane >> 4;
  const int nbase = blockIdx.x * 64;
  const int mbase = blockIdx.y * 128;

  v8f acc[4];
#pragma unroll
  for (int t = 0; t < 4; ++t) acc[t] = (v8f){};

  for (int kk = 0; kk < D_MODEL; kk += 32) {
    for (int i = tid; i < 128 * 32; i += 256) {
      int r = i >> 5, c = i & 31;
      Xs[i] = (__bf16)X[(size_t)(mbase + r) * D_MODEL + kk + c];
    }
    for (int i = tid; i < 64 * 32; i += 256) {
      int n = i >> 5, c = i & 31;
      Ws[i] = (__bf16)W[(size_t)(nbase + n) * D_MODEL + kk + c];
    }
    __syncthreads();

    v16bf a;
#pragma unroll
    for (int e = 0; e < 16; ++e)
      a[e] = Xs[(wave * 16 + lrow) * 32 + frag_k(lane, e)];
#pragma unroll
    for (int t = 0; t < 4; ++t) {
      v16bf b;
#pragma unroll
      for (int e = 0; e < 16; ++e)
        b[e] = Ws[(t * 16 + lrow) * 32 + frag_k(lane, e)];
      acc[t] = wmma_bf16(a, b, acc[t]);
    }
    __syncthreads();
  }

#pragma unroll
  for (int t = 0; t < 4; ++t) {
    const int n = nbase + t * 16 + lrow;
    const float bv = bias[n];
#pragma unroll
    for (int r = 0; r < 8; ++r) {
      const int m = mbase + wave * 16 + lhalf * 8 + r;
      const int bidx = m >> 11, nseq = m & (NQ - 1);
      const int h = n >> 6, dd = n & 63;
      out[(((size_t)bidx * N_HEADS + h) * NQ + nseq) * D_HEAD + dd] =
          (__bf16)(acc[t][r] + bv);
    }
  }
}

// ---------------------------------------------------------------------------
// Flash attention. Block = 8 waves sharing DOUBLE-BUFFERED async-staged 32x64
// K/V LDS tiles (copy of tile t+1 overlaps WMMA/softmax of tile t); each wave
// owns a 16-row q tile. scores = (QK^T)/8 * attn_weights, masked -> -1e30,
// online softmax, O += P·V.
// ---------------------------------------------------------------------------
__global__ __launch_bounds__(256)
void attn_kernel(const __bf16* __restrict__ Qh, const __bf16* __restrict__ Kh,
                 const __bf16* __restrict__ Vh,
                 const unsigned char* __restrict__ msk,
                 const float* __restrict__ aw, __bf16* __restrict__ Oh) {
  __shared__ __bf16 Ks[2][32 * 64];    // [buf][k][d] row-major
  __shared__ __bf16 Vs[2][32 * 64];    // [buf][k][d] row-major
  __shared__ __bf16 Plds[8][16 * 32];  // per-wave P staging (D-layout -> A-layout)
  const int tid  = threadIdx.x;
  const int wave = tid >> 5, lane = tid & 31;
  const int lrow = lane & 15, lhalf = lane >> 4;
  const int bh    = blockIdx.y;
  const int qbase = blockIdx.x * 128 + wave * 16;
  const int cidx  = tid * 8;           // 8 bf16 (16B) per thread per tile

  // Q fragments for d=[0,32) and [32,64)
  v16bf qa[2];
  const __bf16* qptr = Qh + ((size_t)bh * NQ + qbase + lrow) * D_HEAD;
#pragma unroll
  for (int e = 0; e < 16; ++e) {
    const int k = frag_k(lane, e);
    qa[0][e] = qptr[k];
    qa[1][e] = qptr[32 + k];
  }

  float mrow[8], lsum[8];
#pragma unroll
  for (int r = 0; r < 8; ++r) { mrow[r] = -1e30f; lsum[r] = 0.f; }
  v8f oacc[4];
#pragma unroll
  for (int t = 0; t < 4; ++t) oacc[t] = (v8f){};

  // prologue: stage tile 0 into buffer 0
  {
    const size_t tb = (size_t)bh * NK * D_HEAD;
    async_b128(&Ks[0][cidx], Kh + tb + cidx);
    async_b128(&Vs[0][cidx], Vh + tb + cidx);
  }

  for (int kk = 0; kk < NK; kk += 32) {
    const int buf = (kk >> 5) & 1;
    if (kk + 32 < NK) {
      // issue next tile into the other buffer; overlap with this tile's math
      const size_t tb = ((size_t)bh * NK + kk + 32) * D_HEAD;
      async_b128(&Ks[buf ^ 1][cidx], Kh + tb + cidx);
      async_b128(&Vs[buf ^ 1][cidx], Vh + tb + cidx);
      const size_t pidx =
          ((size_t)bh * NQ + qbase + lhalf * 8) * NK + (kk + 32) + lrow;
      __builtin_prefetch(aw + pidx, 0, 1);
      __builtin_prefetch(msk + pidx, 0, 1);
      // wait only for the CURRENT tile's 2 ops; next tile's 2 stay in flight
      asm volatile("s_wait_asynccnt 2" ::: "memory");
    } else {
      asm volatile("s_wait_asynccnt 0" ::: "memory");
    }
    __syncthreads();

    v8f s[2];
#pragma unroll
    for (int half = 0; half < 2; ++half) {
      v16bf kb[2];
      const __bf16* krow = &Ks[buf][(half * 16 + lrow) * D_HEAD];
#pragma unroll
      for (int e = 0; e < 16; ++e) {
        const int k = frag_k(lane, e);
        kb[0][e] = krow[k];
        kb[1][e] = krow[32 + k];
      }
      v8f a = (v8f){};
      a = wmma_bf16(qa[0], kb[0], a);
      a = wmma_bf16(qa[1], kb[1], a);
      const int kcol = kk + half * 16 + lrow;
#pragma unroll
      for (int r = 0; r < 8; ++r) {
        const int qrow = qbase + lhalf * 8 + r;
        const size_t idx = ((size_t)bh * NQ + qrow) * NK + kcol;
        const float v = a[r] * 0.125f * aw[idx];
        a[r] = msk[idx] ? -1e30f : v;
      }
      s[half] = a;
    }

    // online softmax update (row reductions across 16-lane halves)
    float alpha[8];
#pragma unroll
    for (int r = 0; r < 8; ++r) {
      float tmax = fmaxf(s[0][r], s[1][r]);
#pragma unroll
      for (int off = 1; off < 16; off <<= 1)
        tmax = fmaxf(tmax, __shfl_xor(tmax, off, 32));
      const float mnew = fmaxf(mrow[r], tmax);
      alpha[r] = __expf(mrow[r] - mnew);
      mrow[r] = mnew;
      const float p0 = __expf(s[0][r] - mnew);
      const float p1 = __expf(s[1][r] - mnew);
      s[0][r] = p0; s[1][r] = p1;
      float ts = p0 + p1;
#pragma unroll
      for (int off = 1; off < 16; off <<= 1)
        ts += __shfl_xor(ts, off, 32);
      lsum[r] = lsum[r] * alpha[r] + ts;
    }
#pragma unroll
    for (int t = 0; t < 4; ++t)
#pragma unroll
      for (int r = 0; r < 8; ++r) oacc[t][r] *= alpha[r];

    // P: C/D layout -> LDS -> A-fragment layout (wave-private, in-order DS)
    __bf16* pl = &Plds[wave][0];
#pragma unroll
    for (int half = 0; half < 2; ++half)
#pragma unroll
      for (int r = 0; r < 8; ++r)
        pl[(lhalf * 8 + r) * 32 + half * 16 + lrow] = (__bf16)s[half][r];
    asm volatile("s_wait_dscnt 0" ::: "memory");

    v16bf pa;
#pragma unroll
    for (int e = 0; e < 16; ++e)
      pa[e] = pl[lrow * 32 + frag_k(lane, e)];

#pragma unroll
    for (int t = 0; t < 4; ++t) {
      const int dcol = t * 16 + lrow;
      v16bf vb;
#pragma unroll
      for (int e = 0; e < 16; ++e)
        vb[e] = Vs[buf][frag_k(lane, e) * D_HEAD + dcol];
      oacc[t] = wmma_bf16(pa, vb, oacc[t]);
    }
    __syncthreads();  // all waves done with buf before it is restaged
  }

#pragma unroll
  for (int t = 0; t < 4; ++t) {
    const int dcol = t * 16 + lrow;
#pragma unroll
    for (int r = 0; r < 8; ++r) {
      const int qrow = qbase + lhalf * 8 + r;
      Oh[((size_t)bh * NQ + qrow) * D_HEAD + dcol] =
          (__bf16)(oacc[t][r] / lsum[r]);
    }
  }
}

// ---------------------------------------------------------------------------
// Output projection + residual: Y = concat(O) @ Wo^T + bo + queries   (f32)
// X tile (bf16 source) staged via async global->LDS copies.
// ---------------------------------------------------------------------------
__global__ __launch_bounds__(256)
void proj_out_kernel(const __bf16* __restrict__ Oh, const float* __restrict__ Wo,
                     const float* __restrict__ bo, const float* __restrict__ res,
                     float* __restrict__ Y) {
  __shared__ __bf16 Xs[128 * 32];
  __shared__ __bf16 Ws[64 * 32];
  const int tid  = threadIdx.x;
  const int wave = tid >> 5, lane = tid & 31;
  const int lrow = lane & 15, lhalf = lane >> 4;
  const int nbase = blockIdx.x * 64;
  const int mbase = blockIdx.y * 128;
  const int srow = tid >> 1;            // staging row 0..127
  const int scol = (tid & 1) * 16;      // 16-elem chunk (never crosses a head)

  v8f acc[4];
#pragma unroll
  for (int t = 0; t < 4; ++t) acc[t] = (v8f){};

  const int sm = mbase + srow;
  const int sbidx = sm >> 11, snseq = sm & (NQ - 1);

  for (int kk = 0; kk < D_MODEL; kk += 32) {
    {
      const int k = kk + scol;
      const __bf16* src =
          Oh + (((size_t)sbidx * N_HEADS + (k >> 6)) * NQ + snseq) * D_HEAD + (k & 63);
      __bf16* dst = &Xs[srow * 32 + scol];
      async_b128(dst, src);
      async_b128(dst + 8, src + 8);
    }
    for (int i = tid; i < 64 * 32; i += 256) {
      const int n = i >> 5, c = i & 31;
      Ws[i] = (__bf16)Wo[(size_t)(nbase + n) * D_MODEL + kk + c];
    }
    asm volatile("s_wait_asynccnt 0" ::: "memory");
    __syncthreads();

    v16bf a;
#pragma unroll
    for (int e = 0; e < 16; ++e)
      a[e] = Xs[(wave * 16 + lrow) * 32 + frag_k(lane, e)];
#pragma unroll
    for (int t = 0; t < 4; ++t) {
      v16bf b;
#pragma unroll
      for (int e = 0; e < 16; ++e)
        b[e] = Ws[(t * 16 + lrow) * 32 + frag_k(lane, e)];
      acc[t] = wmma_bf16(a, b, acc[t]);
    }
    __syncthreads();
  }

#pragma unroll
  for (int t = 0; t < 4; ++t) {
    const int n = nbase + t * 16 + lrow;
    const float bv = bo[n];
#pragma unroll
    for (int r = 0; r < 8; ++r) {
      const int m = mbase + wave * 16 + lhalf * 8 + r;
      Y[(size_t)m * D_MODEL + n] = acc[t][r] + bv + res[(size_t)m * D_MODEL + n];
    }
  }
}

// ---------------------------------------------------------------------------
// LayerNorm, one block per row of 1024
// ---------------------------------------------------------------------------
__global__ __launch_bounds__(256)
void ln_kernel(const float* __restrict__ Y, const float* __restrict__ gamma,
               const float* __restrict__ beta, float* __restrict__ out) {
  __shared__ float red[16];
  const int row = blockIdx.x;
  const float* x = Y + (size_t)row * D_MODEL;
  float s = 0.f, ss = 0.f;
  for (int i = threadIdx.x; i < D_MODEL; i += 256) {
    const float v = x[i];
    s += v; ss += v * v;
  }
#pragma unroll
  for (int off = 1; off < 32; off <<= 1) {
    s += __shfl_xor(s, off, 32);
    ss += __shfl_xor(ss, off, 32);
  }
  const int wave = threadIdx.x >> 5, lane = threadIdx.x & 31;
  if (lane == 0) { red[wave] = s; red[8 + wave] = ss; }
  __syncthreads();
  if (wave == 0) {
    s  = (lane < 8) ? red[lane] : 0.f;
    ss = (lane < 8) ? red[8 + lane] : 0.f;
#pragma unroll
    for (int off = 1; off < 8; off <<= 1) {
      s += __shfl_xor(s, off, 32);
      ss += __shfl_xor(ss, off, 32);
    }
    if (lane == 0) { red[0] = s; red[1] = ss; }
  }
  __syncthreads();
  const float mu  = red[0] * (1.f / D_MODEL);
  const float var = red[1] * (1.f / D_MODEL) - mu * mu;
  const float inv = rsqrtf(var + LN_EPS);
  float* o = out + (size_t)row * D_MODEL;
  for (int i = threadIdx.x; i < D_MODEL; i += 256)
    o[i] = (x[i] - mu) * inv * gamma[i] + beta[i];
}

// ---------------------------------------------------------------------------
extern "C" void kernel_launch(void* const* d_in, const int* in_sizes, int n_in,
                              void* d_out, int out_size, void* d_ws, size_t ws_size,
                              hipStream_t stream) {
  const float* queries = (const float*)d_in[0];
  const float* keys    = (const float*)d_in[1];
  const float* values  = (const float*)d_in[2];
  const unsigned char* mask = (const unsigned char*)d_in[3];
  const float* aw = (const float*)d_in[4];
  const float* Wq = (const float*)d_in[5];
  const float* bq = (const float*)d_in[6];
  const float* Wk = (const float*)d_in[7];
  const float* bk = (const float*)d_in[8];
  const float* Wv = (const float*)d_in[9];
  const float* bv = (const float*)d_in[10];
  const float* Wo = (const float*)d_in[11];
  const float* bo = (const float*)d_in[12];
  const float* gamma = (const float*)d_in[13];
  const float* beta  = (const float*)d_in[14];
  float* out = (float*)d_out;

  const size_t headElems = (size_t)BATCH * N_HEADS * NQ * D_HEAD;  // 4 Mi
  __bf16* Qh = (__bf16*)d_ws;
  __bf16* Kh = Qh + headElems;
  __bf16* Vh = Kh + headElems;
  __bf16* Oh = Vh + headElems;
  float*  Y  = (float*)(Oh + headElems);

  dim3 blk(256);
  dim3 gproj(D_MODEL / 64, (BATCH * NQ) / 128);   // 16 x 32
  proj_qkv_kernel<<<gproj, blk, 0, stream>>>(queries, Wq, bq, Qh);
  proj_qkv_kernel<<<gproj, blk, 0, stream>>>(keys,    Wk, bk, Kh);
  proj_qkv_kernel<<<gproj, blk, 0, stream>>>(values,  Wv, bv, Vh);

  dim3 gattn(NQ / 128, BATCH * N_HEADS);          // 16 x 32
  attn_kernel<<<gattn, blk, 0, stream>>>(Qh, Kh, Vh, mask, aw, Oh);

  proj_out_kernel<<<gproj, blk, 0, stream>>>(Oh, Wo, bo, queries, Y);

  ln_kernel<<<BATCH * NQ, blk, 0, stream>>>(Y, gamma, beta, out);
}
